// MambaDecoder_66675072303358
// MI455X (gfx1250) — matrix operand order repaired and checked by reading, not compiled, not run on hardware
//
#include <hip/hip_runtime.h>
#include <hip/hip_bf16.h>

typedef __attribute__((ext_vector_type(16))) _Float16 v16h;
typedef __attribute__((ext_vector_type(8)))  float    v8f;

namespace {
constexpr int NB = 4, LL = 4096, M = NB * LL;
constexpr int DIN = 64, DM = 128, DC = 32, DSTATE = 16, DD = 8, KER_ = 4;
constexpr int NCH = 16, CHL = LL / NCH;    // 16 chunks of 256 for the parallel scan
constexpr int NLAYER = 4;
}

// ---------------------------------------------------------------------------
// WMMA GEMM, NT output tiles per wave, fully unrolled over K.
//   A  : f16 row-major [M x K]  (per-lane fragment = two contiguous 16B loads)
//   Bp : f16 packed fragments [(K/32) x (N/16) x 32 lanes x 16 halves]
//   C  : f32 row-major [M x N]
// All NT B fragments for a k-block are loaded as one clause before the WMMA
// chain so later loads overlap earlier WMMAs. Grid exactly covers all tiles
// -> EXEC all-ones for every v_wmma.
// ---------------------------------------------------------------------------
template <int NT, int K, int N>
__global__ void k_wmma_gemm(const _Float16* __restrict__ A,
                            const _Float16* __restrict__ Bp,
                            float* __restrict__ C) {
  constexpr int NTILES = N >> 4;
  constexpr int NGRP   = NTILES / NT;
  const int lane = threadIdx.x & 31;
  const int wv   = blockIdx.x * (blockDim.x >> 5) + (threadIdx.x >> 5);
  const int tm   = wv / NGRP;
  const int tn0  = (wv % NGRP) * NT;
  const int row  = lane & 15;
  const int hi   = lane >> 4;

  v8f acc[NT];
#pragma unroll
  for (int j = 0; j < NT; ++j) acc[j] = (v8f){};

  const _Float16* Abase = A + (size_t)(tm * 16 + row) * K + hi * 8;
  const _Float16* Bbase = Bp + ((size_t)tn0 * 32 + lane) * 16;
#pragma unroll
  for (int kb = 0; kb < K / 32; ++kb) {
    // ---- stage all fragments for this k-block (one load clause) ----
    v16h af;
    const _Float16* Ap = Abase + kb * 32;
#pragma unroll
    for (int i = 0; i < 8;  ++i) af[i] = Ap[i];        // K = kb*32 + hi*8 + i
#pragma unroll
    for (int i = 8; i < 16; ++i) af[i] = Ap[i + 8];    // K = kb*32 + hi*8 + i + 8
    v16h bf[NT];
#pragma unroll
    for (int j = 0; j < NT; ++j) {
      const _Float16* Bq = Bbase + ((size_t)kb * NTILES + j) * 32 * 16;
#pragma unroll
      for (int i = 0; i < 16; ++i) bf[j][i] = Bq[i];   // two contiguous 16B loads
    }
    // ---- back-to-back WMMA chain ----
#pragma unroll
    for (int j = 0; j < NT; ++j)
      acc[j] = __builtin_amdgcn_wmma_f32_16x16x32_f16(false, af, false, bf[j],
                                                      (short)0, acc[j],
                                                      false, false);
  }
#pragma unroll
  for (int j = 0; j < NT; ++j) {
    float* Cp = C + (size_t)(tm * 16 + hi * 8) * N + (tn0 + j) * 16 + row;
#pragma unroll
    for (int r = 0; r < 8; ++r) Cp[(size_t)r * N] = acc[j][r];
  }
}

// Pack a row-major [K x N] matrix (f32 or f16) into WMMA B-fragment layout.
template <typename T>
__global__ void k_pack_b(const T* __restrict__ src, _Float16* __restrict__ dst,
                         int K, int N) {
  int t = blockIdx.x * blockDim.x + threadIdx.x;
  if (t >= K * N) return;
  int n = t % N, k = t / N;
  int kblk = k >> 5, rem = k & 31, hi = (rem >> 3) & 1, i = (rem & 7) + ((rem >> 4) << 3);
  int tn = n >> 4, lane = hi * 16 + (n & 15);
  int ntiles = N >> 4;
  dst[(((size_t)kblk * ntiles + tn) * 32 + lane) * 16 + i] = (_Float16)src[t];
}

// ---------------------------------------------------------------------------
// Elementwise / setup kernels
// ---------------------------------------------------------------------------
__global__ void k_copy(const float* __restrict__ a, float* __restrict__ b, int n) {
  int t = blockIdx.x * blockDim.x + threadIdx.x;
  if (t < n) b[t] = a[t];
}
__global__ void k_add(float* __restrict__ a, const float* __restrict__ b, int n) {
  int t = blockIdx.x * blockDim.x + threadIdx.x;
  if (t < n) a[t] += b[t];
}
// Concatenated per-layer weight [128 x 48] = [sB(16) | sC(16) | sD1(8) | zeros(8)]
__global__ void k_build_wcat(const float* __restrict__ sB, const float* __restrict__ sC,
                             const float* __restrict__ sD1, _Float16* __restrict__ o) {
  int t = blockIdx.x * blockDim.x + threadIdx.x;            // NLAYER*DM*48
  if (t >= NLAYER * DM * 48) return;
  int n = t % 48, d = (t / 48) % DM, l = t / (48 * DM);
  float v = 0.0f;
  if (n < 16)      v = sB [((size_t)l * DM + d) * DSTATE + n];
  else if (n < 32) v = sC [((size_t)l * DM + d) * DSTATE + (n - 16)];
  else if (n < 40) v = sD1[((size_t)l * DM + d) * DD + (n - 32)];
  o[t] = (_Float16)v;
}
// sD2 padded from [8 x 128] to [32 x 128] (zero rows 8..31) per layer
__global__ void k_build_sd2(const float* __restrict__ sD2, _Float16* __restrict__ o) {
  int t = blockIdx.x * blockDim.x + threadIdx.x;            // NLAYER*32*DM
  if (t >= NLAYER * 32 * DM) return;
  int n = t & (DM - 1), k = (t >> 7) & 31, l = t >> 12;
  o[t] = (k < DD) ? (_Float16)sD2[((size_t)l * DD + k) * DM + n] : (_Float16)0.0f;
}

// rmsnorm(seq)*g then concat with context -> f16 [M, 96]. One wave per token.
__global__ void k_rms_concat(const float* __restrict__ seq, const float* __restrict__ ctx,
                             const float* __restrict__ g, _Float16* __restrict__ out) {
  int lane = threadIdx.x & 31;
  int tok  = blockIdx.x * (blockDim.x >> 5) + (threadIdx.x >> 5);
  const float* xr = seq + (size_t)tok * DIN;
  float x0 = xr[lane], x1 = xr[lane + 32];
  float ss = x0 * x0 + x1 * x1;
#pragma unroll
  for (int off = 16; off; off >>= 1) ss += __shfl_xor(ss, off, 32);
  float r = rsqrtf(ss * (1.0f / DIN) + 1e-8f);
  _Float16* o = out + (size_t)tok * (DIN + DC);
  o[lane]      = (_Float16)(x0 * r * g[lane]);
  o[lane + 32] = (_Float16)(x1 * r * g[lane + 32]);
  o[lane + 64] = (_Float16)(ctx[(size_t)tok * DC + lane]);
}

// Depthwise causal conv (K=4) + bias + SiLU on the 'a' half of in-proj output.
__global__ void k_conv_silu(const float* __restrict__ ab, const float* __restrict__ w,
                            const float* __restrict__ bias, float* __restrict__ ap,
                            _Float16* __restrict__ aph) {
  int t = blockIdx.x * blockDim.x + threadIdx.x;
  int d = t & (DM - 1), tok = t >> 7, l = tok & (LL - 1);
  float acc = bias[d];
#pragma unroll
  for (int k = 0; k < KER_; ++k) {
    int dl = l - (KER_ - 1) + k;
    if (dl >= 0) acc += w[d * KER_ + k] * ab[(size_t)(tok - (KER_ - 1) + k) * (2 * DM) + d];
  }
  float s = acc / (1.0f + __expf(-acc));
  ap[t] = s;
  aph[t] = (_Float16)s;
}

// dd1 (cols 32..39 of bcd) -> f16 padded to 32 cols (zeros beyond 8)
__global__ void k_pad_dd1(const float* __restrict__ bcd, _Float16* __restrict__ o) {
  int t = blockIdx.x * blockDim.x + threadIdx.x;
  int c = t & 31, tok = t >> 5;
  o[t] = (c < DD) ? (_Float16)bcd[(size_t)tok * 48 + 32 + c] : (_Float16)0.0f;
}

// delta = softplus(D + dpre), in place.
__global__ void k_softplus(float* __restrict__ x, const float* __restrict__ Dw) {
  int t = blockIdx.x * blockDim.x + threadIdx.x;
  float v = x[t] + Dw[t & (DM - 1)];
  x[t] = (v > 20.0f) ? v : log1pf(__expf(v));
}

// Scan pass 1: per (b,d,chunk,state) compute cumulative product P and local scan S.
__global__ void k_scan1(const float* __restrict__ delta, const float* __restrict__ ap,
                        const float* __restrict__ bcd, const float* __restrict__ Aw,
                        float* __restrict__ P, float* __restrict__ S) {
  int t = blockIdx.x * blockDim.x + threadIdx.x;
  int s = t & 15, ch = (t >> 4) & (NCH - 1), d = (t >> 8) & (DM - 1), b = t >> 15;
  float ac = Aw[d * DSTATE + s];
  float p = 1.0f, acc = 0.0f;
  int tok0 = b * LL + ch * CHL;
  for (int li = 0; li < CHL; ++li) {
    int tok = tok0 + li;
    float de = delta[(size_t)tok * DM + d];
    float av = __expf(-ac * de);
    acc = av * acc + bcd[(size_t)tok * 48 + s] * de * ap[(size_t)tok * DM + d];
    p *= av;
  }
  int idx = ((b * DM + d) * NCH + ch) * DSTATE + s;
  P[idx] = p; S[idx] = acc;
}

// Scan pass 2: serial combine over the 16 chunks -> initial state per chunk.
__global__ void k_scan_combine(const float* __restrict__ P, const float* __restrict__ S,
                               float* __restrict__ H) {
  int t = blockIdx.x * blockDim.x + threadIdx.x;           // NB*DM*DSTATE
  int s = t & 15, d = (t >> 4) & (DM - 1), b = t >> 11;
  float h = 0.0f;
  for (int ch = 0; ch < NCH; ++ch) {
    int idx = ((b * DM + d) * NCH + ch) * DSTATE + s;
    H[idx] = h;
    h = P[idx] * h + S[idx];
  }
}

// Scan pass 3: replay with correct init, contract with C across the 16 states
// (wave32 shfl_xor reduction), fuse the SiLU gate, emit f16 operand for out-proj.
__global__ void k_scan2(const float* __restrict__ delta, const float* __restrict__ ap,
                        const float* __restrict__ bcd, const float* __restrict__ Aw,
                        const float* __restrict__ Dw, const float* __restrict__ ab2,
                        const float* __restrict__ H, _Float16* __restrict__ gated) {
  int t = blockIdx.x * blockDim.x + threadIdx.x;
  int s = t & 15, ch = (t >> 4) & (NCH - 1), d = (t >> 8) & (DM - 1), b = t >> 15;
  float ac = Aw[d * DSTATE + s];
  float h  = H[((b * DM + d) * NCH + ch) * DSTATE + s];
  float dcoef = Dw[d];
  int tok0 = b * LL + ch * CHL;
  for (int li = 0; li < CHL; ++li) {
    int tok = tok0 + li;
    float de  = delta[(size_t)tok * DM + d];
    float apv = ap[(size_t)tok * DM + d];
    float av  = __expf(-ac * de);
    h = av * h + bcd[(size_t)tok * 48 + s] * de * apv;
    float pr = h * bcd[(size_t)tok * 48 + 16 + s];
#pragma unroll
    for (int off = 8; off; off >>= 1) pr += __shfl_xor(pr, off, 32);
    if (s == 0) {
      float br  = ab2[(size_t)tok * (2 * DM) + DM + d];
      float sil = br / (1.0f + __expf(-br));
      gated[(size_t)tok * DM + d] = (_Float16)((pr + dcoef * apv) * sil);
    }
  }
}

// Mean pool over L: one wave per (b, feature).
__global__ void k_pool(const float* __restrict__ seq, float* __restrict__ pooled) {
  int lane = threadIdx.x & 31;
  int gid  = blockIdx.x * (blockDim.x >> 5) + (threadIdx.x >> 5);   // 0..255
  int b = gid >> 6, j = gid & 63;
  float sum = 0.0f;
  for (int l = lane; l < LL; l += 32) sum += seq[((size_t)b * LL + l) * DIN + j];
#pragma unroll
  for (int off = 16; off; off >>= 1) sum += __shfl_xor(sum, off, 32);
  if (lane == 0) pooled[b * DIN + j] = sum * (1.0f / LL);
}

// Final MLP 64 -> 128 (relu) -> 1 (sigmoid). One block of 128 threads.
__global__ void k_mlp(const float* __restrict__ pooled, const float* __restrict__ W1,
                      const float* __restrict__ b1, const float* __restrict__ W2,
                      const float* __restrict__ b2, float* __restrict__ out) {
  __shared__ float hsh[NB * 2 * DIN];
  int j = threadIdx.x;
  for (int b = 0; b < NB; ++b) {
    float acc = b1[j];
    for (int k = 0; k < DIN; ++k) acc += pooled[b * DIN + k] * W1[k * (2 * DIN) + j];
    hsh[b * (2 * DIN) + j] = fmaxf(acc, 0.0f);
  }
  __syncthreads();
  if (j < NB) {
    float o = b2[0];
    for (int k = 0; k < 2 * DIN; ++k) o += hsh[j * (2 * DIN) + k] * W2[k];
    out[j] = 1.0f / (1.0f + __expf(-o));
  }
}

// ---------------------------------------------------------------------------
extern "C" void kernel_launch(void* const* d_in, const int* in_sizes, int n_in,
                              void* d_out, int out_size, void* d_ws, size_t ws_size,
                              hipStream_t stream) {
  (void)in_sizes; (void)n_in; (void)out_size; (void)ws_size;
  const float* x     = (const float*)d_in[0];
  const float* ctx   = (const float*)d_in[1];
  const float* normg = (const float*)d_in[2];
  const float* inW   = (const float*)d_in[3];
  const float* convw = (const float*)d_in[4];
  const float* convb = (const float*)d_in[5];
  const float* sB    = (const float*)d_in[6];
  const float* sC    = (const float*)d_in[7];
  const float* sD1   = (const float*)d_in[8];
  const float* sD2   = (const float*)d_in[9];
  const float* Aw    = (const float*)d_in[10];
  const float* Dw    = (const float*)d_in[11];
  const float* outW  = (const float*)d_in[12];
  const float* W1    = (const float*)d_in[13];
  const float* b1    = (const float*)d_in[14];
  const float* W2    = (const float*)d_in[15];
  const float* b2    = (const float*)d_in[16];
  float* out = (float*)d_out;

  char* p = (char*)d_ws;
  auto alloc = [&](size_t bytes) -> char* {
    char* r = p; p += (bytes + 255) & ~size_t(255); return r;
  };
  float*    seq    = (float*)   alloc((size_t)M * DIN * 4);
  _Float16* cat    = (_Float16*)alloc((size_t)M * 96 * 2);
  float*    abuf   = (float*)   alloc((size_t)M * 256 * 4);
  float*    apost  = (float*)   alloc((size_t)M * DM * 4);
  _Float16* aposth = (_Float16*)alloc((size_t)M * DM * 2);
  float*    bcd    = (float*)   alloc((size_t)M * 48 * 4);
  _Float16* dd1h   = (_Float16*)alloc((size_t)M * 32 * 2);
  float*    delta  = (float*)   alloc((size_t)M * DM * 4);
  float*    Pc     = (float*)   alloc((size_t)NB * DM * NCH * DSTATE * 4);
  float*    Sc     = (float*)   alloc((size_t)NB * DM * NCH * DSTATE * 4);
  float*    Hc     = (float*)   alloc((size_t)NB * DM * NCH * DSTATE * 4);
  _Float16* gated  = (_Float16*)alloc((size_t)M * DM * 2);
  float*    dy     = (float*)   alloc((size_t)M * DIN * 4);
  float*    pooled = (float*)   alloc((size_t)NB * DIN * 4);
  _Float16* wcath  = (_Float16*)alloc((size_t)NLAYER * DM * 48 * 2);
  _Float16* sd2h   = (_Float16*)alloc((size_t)NLAYER * 32 * DM * 2);
  _Float16* inWp   = (_Float16*)alloc((size_t)NLAYER * 96 * 256 * 2);
  _Float16* wcatp  = (_Float16*)alloc((size_t)NLAYER * DM * 48 * 2);
  _Float16* sd2p   = (_Float16*)alloc((size_t)NLAYER * 32 * DM * 2);
  _Float16* outWp  = (_Float16*)alloc((size_t)NLAYER * DM * DIN * 2);

  // Setup: residual init + weight build/convert/pack into fragment layout.
  k_copy<<<(M * DIN + 255) / 256, 256, 0, stream>>>(x, seq, M * DIN);
  k_build_wcat<<<(NLAYER * DM * 48 + 255) / 256, 256, 0, stream>>>(sB, sC, sD1, wcath);
  k_build_sd2 <<<(NLAYER * 32 * DM + 255) / 256, 256, 0, stream>>>(sD2, sd2h);
  for (int l = 0; l < NLAYER; ++l) {
    k_pack_b<float><<<(96 * 256 + 255) / 256, 256, 0, stream>>>(
        inW + (size_t)l * 96 * 256, inWp + (size_t)l * 96 * 256, 96, 256);
    k_pack_b<_Float16><<<(DM * 48 + 255) / 256, 256, 0, stream>>>(
        wcath + (size_t)l * DM * 48, wcatp + (size_t)l * DM * 48, DM, 48);
    k_pack_b<_Float16><<<(32 * DM + 255) / 256, 256, 0, stream>>>(
        sd2h + (size_t)l * 32 * DM, sd2p + (size_t)l * 32 * DM, 32, DM);
    k_pack_b<float><<<(DM * DIN + 255) / 256, 256, 0, stream>>>(
        outW + (size_t)l * DM * DIN, outWp + (size_t)l * DM * DIN, DM, DIN);
  }

  for (int l = 0; l < NLAYER; ++l) {
    k_rms_concat<<<M / 8, 256, 0, stream>>>(seq, ctx, normg + l * DIN, cat);
    // in-proj: [M,96] x [96,256]
    k_wmma_gemm<4, 96, 256><<<(M / 16) * 4 / 4, 128, 0, stream>>>(
        cat, inWp + (size_t)l * 96 * 256, abuf);
    k_conv_silu<<<M * DM / 256, 256, 0, stream>>>(
        abuf, convw + l * DM * KER_, convb + l * DM, apost, aposth);
    // B|C|dd1 projections: [M,128] x [128,48]
    k_wmma_gemm<3, 128, 48><<<(M / 16) / 4, 128, 0, stream>>>(
        aposth, wcatp + (size_t)l * DM * 48, bcd);
    k_pad_dd1<<<M * 32 / 256, 256, 0, stream>>>(bcd, dd1h);
    // delta pre-activation: [M,32pad] x [32pad,128]
    k_wmma_gemm<4, 32, 128><<<(M / 16) * 2 / 4, 128, 0, stream>>>(
        dd1h, sd2p + (size_t)l * 32 * DM, delta);
    k_softplus<<<M * DM / 256, 256, 0, stream>>>(delta, Dw + l * DM);
    // chunked parallel selective scan
    k_scan1<<<NB * DM * NCH * DSTATE / 256, 256, 0, stream>>>(
        delta, apost, bcd, Aw + l * DM * DSTATE, Pc, Sc);
    k_scan_combine<<<NB * DM * DSTATE / 256, 256, 0, stream>>>(Pc, Sc, Hc);
    k_scan2<<<NB * DM * NCH * DSTATE / 256, 256, 0, stream>>>(
        delta, apost, bcd, Aw + l * DM * DSTATE, Dw + l * DM, abuf, Hc, gated);
    // out-proj: [M,128] x [128,64]
    k_wmma_gemm<4, 128, 64><<<(M / 16) / 4, 128, 0, stream>>>(
        gated, outWp + (size_t)l * DM * DIN, dy);
    k_add<<<M * DIN / 256, 256, 0, stream>>>(seq, dy, M * DIN);
  }
  k_pool<<<32, 256, 0, stream>>>(seq, pooled);
  k_mlp<<<1, 128, 0, stream>>>(pooled, W1, b1, W2, b2, out);
}
